// WQ8Linear_GEMM_13735305412690
// MI455X (gfx1250) — matrix operand order repaired and checked by reading, not compiled, not run on hardware
//
#include <hip/hip_runtime.h>
#include <stdint.h>

typedef __attribute__((ext_vector_type(16))) _Float16     v16h;
typedef __attribute__((ext_vector_type(8)))  _Float16     v8h;
typedef __attribute__((ext_vector_type(2)))  _Float16     h2;
typedef __attribute__((ext_vector_type(8)))  float        v8f;
typedef __attribute__((ext_vector_type(4)))  unsigned int v4u;
typedef __attribute__((ext_vector_type(4)))  int          v4i;

#define M_TOTAL 8192
#define N_TOTAL 11008
#define K_TOTAL 4096
#define GROUPSZ 128

#define BM 128
#define BN 128
#define BK 32
#define NSTEP (K_TOTAL / BK)   // 128
#define LDA 40                 // padded row stride (halves) for A tile
#define LDW 40                 // padded row stride (halves) for W tile
#define A_TILE (BM * LDA)
#define W_TILE (BN * LDW)

#if __has_builtin(__builtin_amdgcn_global_load_async_to_lds_b128)
#define USE_ASYNC_A 1
typedef __attribute__((address_space(1))) v4i* as1_v4i_ptr;
typedef __attribute__((address_space(3))) v4i* as3_v4i_ptr;
#else
#define USE_ASYNC_A 0
#endif

__device__ __forceinline__ void wait_async_zero() {
#if __has_builtin(__builtin_amdgcn_s_wait_asynccnt)
    __builtin_amdgcn_s_wait_asynccnt(0);
#else
    asm volatile("s_wait_asynccnt 0" ::: "memory");
#endif
}

__global__ __launch_bounds__(256) void wq8_gemm_wmma_kernel(
    const _Float16* __restrict__ x,       // [M_TOTAL, K_TOTAL] fp16
    const signed char* __restrict__ w,    // [N_TOTAL, K_TOTAL] int8
    const _Float16* __restrict__ wscale,  // [N_TOTAL, K_TOTAL/GROUPSZ] fp16
    const _Float16* __restrict__ bias,    // [N_TOTAL] fp16
    _Float16* __restrict__ out)           // [M_TOTAL, N_TOTAL] fp16
{
    __shared__ _Float16 lds[2][A_TILE + W_TILE];

    const int tid    = threadIdx.x;
    const int lane   = tid & 31;
    const int wave   = tid >> 5;
    const int wave_m = wave & 1;   // 0..1 -> 64 M-rows each
    const int wave_n = wave >> 1;  // 0..3 -> 32 N-cols each

    const int n0 = blockIdx.x * BN;
    const int m0 = blockIdx.y * BM;

    // ---- staging mapping: 2 threads per tile row, 16 elems each ----
    const int s_row = tid >> 1;         // 0..127
    const int s_col = (tid & 1) * 16;   // 0 or 16

    const _Float16*    xg = x + (size_t)(m0 + s_row) * K_TOTAL + s_col;
    const signed char* wg = w + (size_t)(n0 + s_row) * K_TOTAL + s_col;
    const _Float16*    sg = wscale + (size_t)(n0 + s_row) * (K_TOTAL / GROUPSZ);

    v4u      w_reg;     // 16 int8 weights
    _Float16 s_reg;     // fp16 dequant scale for this (row, group)
#if !USE_ASYNC_A
    v8h a_reg0, a_reg1; // 32B of x row (register-staged fallback)
#endif

    auto load_stage = [&](int step, int buf) {
        const int k0 = step * BK;
#if USE_ASYNC_A
        // Async DMA the A tile straight into the back LDS buffer (ASYNCcnt).
        _Float16* Ad = lds[buf] + s_row * LDA + s_col;
        __builtin_amdgcn_global_load_async_to_lds_b128(
            (as1_v4i_ptr)(v4i*)(xg + k0),
            (as3_v4i_ptr)(v4i*)Ad, 0, 0);
        __builtin_amdgcn_global_load_async_to_lds_b128(
            (as1_v4i_ptr)(v4i*)(xg + k0 + 8),
            (as3_v4i_ptr)(v4i*)(Ad + 8), 0, 0);
#else
        (void)buf;
        a_reg0 = *(const v8h*)(xg + k0);
        a_reg1 = *(const v8h*)(xg + k0 + 8);
#endif
        w_reg = *(const v4u*)(wg + k0);
        s_reg = sg[step >> 2];   // GROUPSZ/BK = 4 steps per group
    };

    auto store_stage = [&](int buf) {
        _Float16* As = lds[buf];
        _Float16* Ws = lds[buf] + A_TILE;
#if !USE_ASYNC_A
        *(v8h*)(As + s_row * LDA + s_col)     = a_reg0;
        *(v8h*)(As + s_row * LDA + s_col + 8) = a_reg1;
#else
        (void)As;
#endif
        // Branchless int8 -> fp16 dequant:
        //   u8 = s8 ^ 0x80;  as_f16(0x6400 | u8) == 1024 + u8 (exact)
        //   w_f16 = (as_f16(0x6400|u8) - 1152) * scale   (sub is exact in fp16)
        h2 sc2; sc2[0] = s_reg;            sc2[1] = s_reg;
        h2 kb;  kb[0]  = (_Float16)1152.f; kb[1]  = (_Float16)1152.f;
        v8h outv[2];
        #pragma unroll
        for (int j = 0; j < 4; ++j) {
            uint32_t ux  = w_reg[j] ^ 0x80808080u;
            uint32_t p01 = __builtin_amdgcn_perm(0x64646464u, ux, 0x04010400u);
            uint32_t p23 = __builtin_amdgcn_perm(0x64646464u, ux, 0x04030402u);
            h2 h01 = (__builtin_bit_cast(h2, p01) - kb) * sc2;
            h2 h23 = (__builtin_bit_cast(h2, p23) - kb) * sc2;
            const int base = (j & 1) * 4;
            outv[j >> 1][base + 0] = h01[0];
            outv[j >> 1][base + 1] = h01[1];
            outv[j >> 1][base + 2] = h23[0];
            outv[j >> 1][base + 3] = h23[1];
        }
        *(v8h*)(Ws + s_row * LDW + s_col)     = outv[0];
        *(v8h*)(Ws + s_row * LDW + s_col + 8) = outv[1];
    };

    // ---- WMMA fragment lane mapping (per ISA 7.12.2, wave32) ----
    const int r   = lane & 15;   // row (A) / column (B,C,D)
    const int hlf = lane >> 4;   // half-wave selector

    v8f acc[4][2] = {};          // 4 M-subtiles x 2 N-subtiles, f32 16x16 C/D

    load_stage(0, 0);
    store_stage(0);
#if USE_ASYNC_A
    wait_async_zero();
#endif
    __syncthreads();

    for (int step = 0; step < NSTEP; ++step) {
        const int nxt = step + 1;
        // Back buffer (nxt&1) was last read in step-1; the barrier that ended
        // step-1 makes it safe to start filling it now (async + ds stores).
        if (nxt < NSTEP) load_stage(nxt, nxt & 1);

        const _Float16* As = lds[step & 1];
        const _Float16* Ws = lds[step & 1] + A_TILE;

        // B fragments: 32x16 f16. lane (r, hlf): col=r, elems 0..7 -> K = hlf*8 .. hlf*8+7, elems 8..15 -> K = 16 + hlf*8 .. 16 + hlf*8+7
        v16h bfrag[2];
        #pragma unroll
        for (int ni = 0; ni < 2; ++ni) {
            const _Float16* p = Ws + (wave_n * 32 + ni * 16 + r) * LDW + hlf * 8;
            v8h b0 = *(const v8h*)p;
            v8h b1 = *(const v8h*)(p + 16);
            bfrag[ni] = __builtin_shufflevector(b0, b1,
                0,1,2,3,4,5,6,7,8,9,10,11,12,13,14,15);
        }

        // A fragments: 16x32 f16. lane (r, hlf): row=r,
        //   elems 0..7  -> K = hlf*8 .. hlf*8+7
        //   elems 8..15 -> K = 16 + hlf*8 .. 16 + hlf*8+7
        #pragma unroll
        for (int mi = 0; mi < 4; ++mi) {
            const _Float16* p = As + (wave_m * 64 + mi * 16 + r) * LDA + hlf * 8;
            v8h a0 = *(const v8h*)p;
            v8h a1 = *(const v8h*)(p + 16);
            v16h afrag = __builtin_shufflevector(a0, a1,
                0,1,2,3,4,5,6,7,8,9,10,11,12,13,14,15);
            #pragma unroll
            for (int ni = 0; ni < 2; ++ni) {
                acc[mi][ni] = __builtin_amdgcn_wmma_f32_16x16x32_f16(
                    /*neg_a=*/false, afrag,
                    /*neg_b=*/false, bfrag[ni],
                    /*c_mod=*/(short)0, acc[mi][ni],
                    /*reuse_a=*/false, /*reuse_b=*/false);
            }
        }

        if (nxt < NSTEP) {
            store_stage(nxt & 1);
#if USE_ASYNC_A
            wait_async_zero();   // A tile for the back buffer is in LDS
#endif
        }
        __syncthreads();
    }

    // ---- epilogue: f32 acc -> fp16, fp16 bias add; stream out (NT) ----
    #pragma unroll
    for (int ni = 0; ni < 2; ++ni) {
        const int col = n0 + wave_n * 32 + ni * 16 + r;
        const _Float16 bv = bias[col];
        #pragma unroll
        for (int mi = 0; mi < 4; ++mi) {
            const int rowbase = m0 + wave_m * 64 + mi * 16 + hlf * 8;
            #pragma unroll
            for (int j = 0; j < 8; ++j) {
                const int row = rowbase + j;
                _Float16 v = (_Float16)acc[mi][ni][j] + bv;
                __builtin_nontemporal_store(v, &out[(size_t)row * N_TOTAL + col]);
            }
        }
    }
}

extern "C" void kernel_launch(void* const* d_in, const int* in_sizes, int n_in,
                              void* d_out, int out_size, void* d_ws, size_t ws_size,
                              hipStream_t stream) {
    const _Float16*    x  = (const _Float16*)d_in[0];
    const signed char* w  = (const signed char*)d_in[1];
    const _Float16*    ws = (const _Float16*)d_in[2];
    const _Float16*    b  = (const _Float16*)d_in[3];
    _Float16*          o  = (_Float16*)d_out;

    dim3 grid(N_TOTAL / BN, M_TOTAL / BM);   // 86 x 64
    wq8_gemm_wmma_kernel<<<grid, 256, 0, stream>>>(x, w, ws, b, o);
}
